// EGNNForResidueIdentity_25821343383882
// MI455X (gfx1250) — matrix operand order repaired and checked by reading, not compiled, not run on hardware
//
#include <hip/hip_runtime.h>

// ---------------- problem constants ----------------
#define N_NODES 50000
#define N_EDGES 800000
#define F_INF   32
#define HDIM    128
#define OUT_DIM 20
#define L_LAYERS 4

// ---------------- WMMA types / helpers ----------------
typedef __attribute__((ext_vector_type(16))) __bf16 v16bf;
typedef __attribute__((ext_vector_type(8)))  float  v8f;

union Frag { unsigned int u[8]; v16bf v; };

__device__ __forceinline__ unsigned short f2bf(float f) {
  unsigned int u = __builtin_bit_cast(unsigned int, f);
  u += 0x7FFFu + ((u >> 16) & 1u);           // round to nearest even
  return (unsigned short)(u >> 16);
}
__device__ __forceinline__ unsigned int pack_bf2(float a, float b) {
  return (unsigned int)f2bf(a) | ((unsigned int)f2bf(b) << 16);
}
__device__ __forceinline__ float bf2f(unsigned short u) {
  unsigned int x = ((unsigned int)u) << 16;
  return __builtin_bit_cast(float, x);
}
__device__ __forceinline__ float silu_f(float x) { return x / (1.0f + __expf(-x)); }
__device__ __forceinline__ float sigm_f(float x) { return 1.0f / (1.0f + __expf(-x)); }

__device__ __forceinline__ v8f wmma_bf16(const Frag& a, const Frag& b, v8f c) {
  // D = A(16x32 bf16) x B(32x16 bf16) + C(16x16 f32)
  return __builtin_amdgcn_wmma_f32_16x16x32_bf16(
      false, a.v, false, b.v, (short)0, c, false, false);
}

// wave-local LDS write->read visibility (cross-lane, same wave)
__device__ __forceinline__ void lds_fence() {
  __builtin_amdgcn_wave_barrier();
  asm volatile("s_wait_dscnt 0" ::: "memory");
  __builtin_amdgcn_wave_barrier();
}

// A-fragment K base per VGPR (16-bit A 16x32 layout), add 8*half
__device__ __forceinline__ int kbA(int v) { return (v < 4) ? (2 * v) : (16 + 2 * (v - 4)); }

__device__ __forceinline__ void load_bfrag(Frag& b, const unsigned int* p) {
  const uint4* p4 = (const uint4*)p;
  uint4 q0 = p4[0], q1 = p4[1];
  b.u[0] = q0.x; b.u[1] = q0.y; b.u[2] = q0.z; b.u[3] = q0.w;
  b.u[4] = q1.x; b.u[5] = q1.y; b.u[6] = q1.z; b.u[7] = q1.w;
}

__device__ __forceinline__ v8f v8f_zero() {
  v8f z;
#pragma unroll
  for (int i = 0; i < 8; ++i) z[i] = 0.0f;
  return z;
}

// ---------------- small utility kernels ----------------
__global__ void zero_f32(float* p, long n) {
  long i = (long)blockIdx.x * blockDim.x + threadIdx.x;
  if (i < n) p[i] = 0.0f;
}
__global__ void copy_f32(const float* s, float* d, long n) {
  long i = (long)blockIdx.x * blockDim.x + threadIdx.x;
  if (i < n) d[i] = s[i];
}
__global__ void count_kernel(const int* __restrict__ edges, float* __restrict__ cnt) {
  int e = blockIdx.x * blockDim.x + threadIdx.x;
  if (e < N_EDGES) atomicAdd(&cnt[edges[e]], 1.0f);
}

// h_bf[n][c] = bf16( h[n] @ emb_in_w + emb_in_b )
__global__ void embed_kernel(const float* __restrict__ h, const float* __restrict__ w,
                             const float* __restrict__ b, unsigned short* __restrict__ h_bf) {
  int i = blockIdx.x * blockDim.x + threadIdx.x;
  if (i >= N_NODES * HDIM) return;
  int n = i >> 7, c = i & 127;
  float acc = b[c];
#pragma unroll 8
  for (int k = 0; k < F_INF; ++k) acc += h[n * F_INF + k] * w[k * HDIM + c];
  h_bf[i] = f2bf(acc);
}

// Swizzle W[L][K][128] (f32) into WMMA bf16 B-fragment order.
// dst dword d = l*Kt*2048 + kt*2048 + nt*256 + lane*8 + v
// element (k, n): n = nt*16 + (lane&15); k = kt*32 + kbA(v) + 8*(lane>>4) (+0/+1 packed)
__global__ void swizzle_w(const float* __restrict__ W, unsigned int* __restrict__ dst,
                          int Kt, int layer_stride) {
  int d = blockIdx.x * blockDim.x + threadIdx.x;
  int per_layer = Kt * 2048;
  if (d >= L_LAYERS * per_layer) return;
  int l = d / per_layer;
  int r = d % per_layer;
  int v = r & 7;
  int lane = (r >> 3) & 31;
  int nt = (r >> 8) & 7;
  int kt = r >> 11;
  int n = nt * 16 + (lane & 15);
  int k = kt * 32 + kbA(v) + 8 * (lane >> 4);
  const float* base = W + (size_t)l * layer_stride;
  dst[d] = pack_bf2(base[k * HDIM + n], base[(k + 1) * HDIM + n]);
}

// ---------------- edge kernel: one wave per 16-edge tile ----------------
__global__ void __launch_bounds__(32) edge_kernel(
    const int* __restrict__ edges, const unsigned short* __restrict__ h_bf,
    const float* __restrict__ x_cur,
    const unsigned int* __restrict__ wf_e1, const unsigned int* __restrict__ wf_e2,
    const unsigned int* __restrict__ wf_c1,
    const float* __restrict__ ew1_full,  // [L][257][128] (for radial row 256)
    const float* __restrict__ eb1, const float* __restrict__ eb2,
    const float* __restrict__ attw, const float* __restrict__ attb,
    const float* __restrict__ cb1, const float* __restrict__ cw2,
    float* __restrict__ agg_m, float* __restrict__ agg_x, int l) {
  __shared__ int lrow[16];
  __shared__ int lcol[16];
  __shared__ float ldx[16], ldy[16], ldz[16], lrad[16];
  __shared__ __align__(16) unsigned short mtile[16 * HDIM];

  const int lane = threadIdx.x;
  const int lane15 = lane & 15;
  const int half = lane >> 4;
  const int e0 = blockIdx.x * 16;

  if (lane < 16) {
    int e = e0 + lane;
    int r = edges[e];
    int c = edges[N_EDGES + e];
    float dx = x_cur[r * 3 + 0] - x_cur[c * 3 + 0];
    float dy = x_cur[r * 3 + 1] - x_cur[c * 3 + 1];
    float dz = x_cur[r * 3 + 2] - x_cur[c * 3 + 2];
    lrow[lane] = r; lcol[lane] = c;
    ldx[lane] = dx; ldy[lane] = dy; ldz[lane] = dz;
    lrad[lane] = dx * dx + dy * dy + dz * dz;
  }
  lds_fence();

  const int nr = lrow[lane15];
  const int nc = lcol[lane15];
  const unsigned short* rowp = h_bf + (size_t)nr * HDIM;
  const unsigned short* colp = h_bf + (size_t)nc * HDIM;

  const unsigned int* w1l = wf_e1 + (size_t)l * (8 * 2048);
  const unsigned int* w2l = wf_e2 + (size_t)l * (4 * 2048);
  const unsigned int* c1l = wf_c1 + (size_t)l * (4 * 2048);
  __builtin_prefetch(w1l, 0, 3);
  __builtin_prefetch(w2l, 0, 3);
  __builtin_prefetch(c1l, 0, 3);

  v8f acc[8];
#pragma unroll
  for (int nt = 0; nt < 8; ++nt) acc[nt] = v8f_zero();

  // ---- GEMM1: [h_row | h_col] @ W1  (K = 256, 8 K-tiles)
#pragma unroll
  for (int kt = 0; kt < 8; ++kt) {
    const unsigned short* src = (kt < 4) ? (rowp + kt * 32) : (colp + (kt - 4) * 32);
    Frag a;
#pragma unroll
    for (int v = 0; v < 8; ++v)
      a.u[v] = *(const unsigned int*)(src + kbA(v) + 8 * half);
#pragma unroll
    for (int nt = 0; nt < 8; ++nt) {
      Frag b; load_bfrag(b, w1l + (size_t)((kt * 8 + nt) * 32 + lane) * 8);
      acc[nt] = wmma_bf16(a, b, acc[nt]);
    }
  }
  // radial rank-1 (K row 256) + bias + SiLU -> LDS bf16 tile
  float radv[8];
#pragma unroll
  for (int v = 0; v < 8; ++v) radv[v] = lrad[v + 8 * half];
#pragma unroll
  for (int nt = 0; nt < 8; ++nt) {
    int col = nt * 16 + lane15;
    float wlast = ew1_full[(size_t)l * 257 * HDIM + 256 * HDIM + col];
    float bv = eb1[l * HDIM + col];
#pragma unroll
    for (int v = 0; v < 8; ++v) {
      float f = silu_f(acc[nt][v] + radv[v] * wlast + bv);
      mtile[(v + 8 * half) * HDIM + col] = f2bf(f);
    }
  }
  lds_fence();

  // ---- GEMM2: m2 = silu(m1 @ W2 + b2), then attention gate
#pragma unroll
  for (int nt = 0; nt < 8; ++nt) acc[nt] = v8f_zero();
#pragma unroll
  for (int kt = 0; kt < 4; ++kt) {
    Frag a;
#pragma unroll
    for (int v = 0; v < 8; ++v)
      a.u[v] = *(const unsigned int*)&mtile[lane15 * HDIM + kt * 32 + kbA(v) + 8 * half];
#pragma unroll
    for (int nt = 0; nt < 8; ++nt) {
      Frag b; load_bfrag(b, w2l + (size_t)((kt * 8 + nt) * 32 + lane) * 8);
      acc[nt] = wmma_bf16(a, b, acc[nt]);
    }
  }
  float p[8];
#pragma unroll
  for (int v = 0; v < 8; ++v) p[v] = 0.0f;
#pragma unroll
  for (int nt = 0; nt < 8; ++nt) {
    int col = nt * 16 + lane15;
    float bv = eb2[l * HDIM + col];
    float aw = attw[l * HDIM + col];
#pragma unroll
    for (int v = 0; v < 8; ++v) {
      float f = silu_f(acc[nt][v] + bv);
      acc[nt][v] = f;
      p[v] += f * aw;
    }
  }
#pragma unroll
  for (int v = 0; v < 8; ++v)
    for (int mask = 1; mask < 16; mask <<= 1) p[v] += __shfl_xor(p[v], mask, 32);
  float ab = attb[l];
  float gate[8];
#pragma unroll
  for (int v = 0; v < 8; ++v) gate[v] = sigm_f(p[v] + ab);

  // gated m -> LDS (for coord MLP) + agg_m atomics (segment sum over row)
  int rn[8];
#pragma unroll
  for (int v = 0; v < 8; ++v) rn[v] = lrow[v + 8 * half];
#pragma unroll
  for (int nt = 0; nt < 8; ++nt) {
    int col = nt * 16 + lane15;
#pragma unroll
    for (int v = 0; v < 8; ++v) {
      float f = acc[nt][v] * gate[v];
      mtile[(v + 8 * half) * HDIM + col] = f2bf(f);
      atomicAdd(&agg_m[(size_t)rn[v] * HDIM + col], f);
    }
  }
  lds_fence();

  // ---- GEMM3: coord MLP: t = silu(m @ C1 + cb1); cw = t @ cw2
#pragma unroll
  for (int nt = 0; nt < 8; ++nt) acc[nt] = v8f_zero();
#pragma unroll
  for (int kt = 0; kt < 4; ++kt) {
    Frag a;
#pragma unroll
    for (int v = 0; v < 8; ++v)
      a.u[v] = *(const unsigned int*)&mtile[lane15 * HDIM + kt * 32 + kbA(v) + 8 * half];
#pragma unroll
    for (int nt = 0; nt < 8; ++nt) {
      Frag b; load_bfrag(b, c1l + (size_t)((kt * 8 + nt) * 32 + lane) * 8);
      acc[nt] = wmma_bf16(a, b, acc[nt]);
    }
  }
  float q[8];
#pragma unroll
  for (int v = 0; v < 8; ++v) q[v] = 0.0f;
#pragma unroll
  for (int nt = 0; nt < 8; ++nt) {
    int col = nt * 16 + lane15;
    float bv = cb1[l * HDIM + col];
    float w2v = cw2[l * HDIM + col];
#pragma unroll
    for (int v = 0; v < 8; ++v) q[v] += silu_f(acc[nt][v] + bv) * w2v;
  }
#pragma unroll
  for (int v = 0; v < 8; ++v)
    for (int mask = 1; mask < 16; mask <<= 1) q[v] += __shfl_xor(q[v], mask, 32);

  // trans = diff * cw ; one lane per row does the agg_x atomics
#pragma unroll
  for (int v = 0; v < 8; ++v) {
    if (lane15 == v) {
      int r = v + 8 * half;
      float cwv = q[v];
      int node = rn[v];
      atomicAdd(&agg_x[node * 3 + 0], ldx[r] * cwv);
      atomicAdd(&agg_x[node * 3 + 1], ldy[r] * cwv);
      atomicAdd(&agg_x[node * 3 + 2], ldz[r] * cwv);
    }
  }
}

// ---------------- node kernel: one wave per 16-node tile ----------------
__global__ void __launch_bounds__(32) node_kernel(
    unsigned short* __restrict__ h_bf, float* __restrict__ x_cur,
    float* __restrict__ agg_m, float* __restrict__ agg_x,
    const float* __restrict__ cnt,
    const unsigned int* __restrict__ wf_n1, const unsigned int* __restrict__ wf_n2,
    const float* __restrict__ nb1, const float* __restrict__ nb2, int l) {
  __shared__ __align__(16) unsigned short mtile[16 * HDIM];

  const int lane = threadIdx.x;
  const int lane15 = lane & 15;
  const int half = lane >> 4;
  const int base = blockIdx.x * 16;

  // coordinate update: x += agg_x / max(cnt, 1); zero agg_x for next layer
  if (lane < 16) {
    int n = base + lane;
    float inv = 1.0f / fmaxf(cnt[n], 1.0f);
#pragma unroll
    for (int j = 0; j < 3; ++j) {
      x_cur[n * 3 + j] += agg_x[n * 3 + j] * inv;
      agg_x[n * 3 + j] = 0.0f;
    }
  }

  const unsigned int* w1l = wf_n1 + (size_t)l * (8 * 2048);
  const unsigned int* w2l = wf_n2 + (size_t)l * (4 * 2048);

  const int myrow = base + lane15;
  v8f acc[8];
#pragma unroll
  for (int nt = 0; nt < 8; ++nt) acc[nt] = v8f_zero();

  // ---- GEMM1: [h | agg_m] @ node_w1  (K = 256)
#pragma unroll
  for (int kt = 0; kt < 8; ++kt) {
    Frag a;
    if (kt < 4) {
      const unsigned short* src = h_bf + (size_t)myrow * HDIM + kt * 32;
#pragma unroll
      for (int v = 0; v < 8; ++v)
        a.u[v] = *(const unsigned int*)(src + kbA(v) + 8 * half);
    } else {
      const float* src = agg_m + (size_t)myrow * HDIM + (kt - 4) * 32;
#pragma unroll
      for (int v = 0; v < 8; ++v) {
        int k0 = kbA(v) + 8 * half;
        float2 f2 = *(const float2*)(src + k0);
        a.u[v] = pack_bf2(f2.x, f2.y);
      }
    }
#pragma unroll
    for (int nt = 0; nt < 8; ++nt) {
      Frag b; load_bfrag(b, w1l + (size_t)((kt * 8 + nt) * 32 + lane) * 8);
      acc[nt] = wmma_bf16(a, b, acc[nt]);
    }
  }
#pragma unroll
  for (int nt = 0; nt < 8; ++nt) {
    int col = nt * 16 + lane15;
    float bv = nb1[l * HDIM + col];
#pragma unroll
    for (int v = 0; v < 8; ++v)
      mtile[(v + 8 * half) * HDIM + col] = f2bf(silu_f(acc[nt][v] + bv));
  }
  lds_fence();

  // ---- GEMM2 + residual; also zero agg_m for next layer
#pragma unroll
  for (int nt = 0; nt < 8; ++nt) acc[nt] = v8f_zero();
#pragma unroll
  for (int kt = 0; kt < 4; ++kt) {
    Frag a;
#pragma unroll
    for (int v = 0; v < 8; ++v)
      a.u[v] = *(const unsigned int*)&mtile[lane15 * HDIM + kt * 32 + kbA(v) + 8 * half];
#pragma unroll
    for (int nt = 0; nt < 8; ++nt) {
      Frag b; load_bfrag(b, w2l + (size_t)((kt * 8 + nt) * 32 + lane) * 8);
      acc[nt] = wmma_bf16(a, b, acc[nt]);
    }
  }
#pragma unroll
  for (int nt = 0; nt < 8; ++nt) {
    int col = nt * 16 + lane15;
    float bv = nb2[l * HDIM + col];
#pragma unroll
    for (int v = 0; v < 8; ++v) {
      int node = base + v + 8 * half;
      size_t idx = (size_t)node * HDIM + col;
      float hv = bf2f(h_bf[idx]) + acc[nt][v] + bv;
      h_bf[idx] = f2bf(hv);
      agg_m[idx] = 0.0f;
    }
  }
}

// ---------------- readout ----------------
__global__ void __launch_bounds__(128) colsum_kernel(const unsigned short* __restrict__ h_bf,
                                                     float* __restrict__ hsum) {
  int c = threadIdx.x;          // 128 columns
  int n0 = blockIdx.x * 200;    // 250 blocks x 200 nodes
  float s = 0.0f;
  for (int i = 0; i < 200; ++i) s += bf2f(h_bf[(size_t)(n0 + i) * HDIM + c]);
  atomicAdd(&hsum[c], s);
}

__global__ void __launch_bounds__(128) head_kernel(
    const float* __restrict__ hsum,
    const float* __restrict__ W0, const float* __restrict__ b0,
    const float* __restrict__ W1, const float* __restrict__ b1,
    const float* __restrict__ W2, const float* __restrict__ b2,
    float* __restrict__ out) {
  __shared__ float bufA[128], bufB[128];
  int c = threadIdx.x;
  bufA[c] = hsum[c] * (1.0f / (float)N_NODES);  // mean(h)
  __syncthreads();
  float a = b0[c];
  for (int k = 0; k < 128; ++k) a += bufA[k] * W0[k * 128 + c];  // emb_out
  bufB[c] = a;
  __syncthreads();
  float a2 = b1[c];
  for (int k = 0; k < 128; ++k) a2 += bufB[k] * W1[k * 128 + c];  // head_w1
  a2 = fmaxf(a2, 0.0f);
  bufA[c] = a2;
  __syncthreads();
  if (c < OUT_DIM) {
    float o = b2[c];
    for (int k = 0; k < 128; ++k) o += bufA[k] * W2[k * OUT_DIM + c];  // head_w2
    out[c] = o;
  }
}

// ---------------- host launch ----------------
extern "C" void kernel_launch(void* const* d_in, const int* in_sizes, int n_in,
                              void* d_out, int out_size, void* d_ws, size_t ws_size,
                              hipStream_t stream) {
  (void)in_sizes; (void)n_in; (void)out_size; (void)ws_size;
  const float* h_in      = (const float*)d_in[0];
  const float* x_in      = (const float*)d_in[1];
  const int*   edges     = (const int*)d_in[2];
  const float* emb_in_w  = (const float*)d_in[3];
  const float* emb_in_b  = (const float*)d_in[4];
  const float* emb_out_w = (const float*)d_in[5];
  const float* emb_out_b = (const float*)d_in[6];
  const float* edge_w1   = (const float*)d_in[7];
  const float* edge_b1   = (const float*)d_in[8];
  const float* edge_w2   = (const float*)d_in[9];
  const float* edge_b2   = (const float*)d_in[10];
  const float* att_w     = (const float*)d_in[11];
  const float* att_b     = (const float*)d_in[12];
  const float* coord_w1  = (const float*)d_in[13];
  const float* coord_b1  = (const float*)d_in[14];
  const float* coord_w2  = (const float*)d_in[15];
  const float* node_w1   = (const float*)d_in[16];
  const float* node_b1   = (const float*)d_in[17];
  const float* node_w2   = (const float*)d_in[18];
  const float* node_b2   = (const float*)d_in[19];
  const float* head_w1   = (const float*)d_in[20];
  const float* head_b1   = (const float*)d_in[21];
  const float* head_w2   = (const float*)d_in[22];
  const float* head_b2   = (const float*)d_in[23];

  char* ws = (char*)d_ws;
  size_t off = 0;
  auto take = [&](size_t bytes) -> char* {
    char* p = ws + off;
    off = (off + bytes + 255) & ~(size_t)255;
    return p;
  };
  unsigned short* h_bf = (unsigned short*)take((size_t)N_NODES * HDIM * 2);
  float* agg_m = (float*)take((size_t)N_NODES * HDIM * 4);
  float* agg_x = (float*)take((size_t)N_NODES * 3 * 4);
  float* cnt   = (float*)take((size_t)N_NODES * 4);
  float* x_cur = (float*)take((size_t)N_NODES * 3 * 4);
  float* hsum  = (float*)take(512);
  unsigned int* wf_e1 = (unsigned int*)take((size_t)L_LAYERS * 8 * 2048 * 4);
  unsigned int* wf_e2 = (unsigned int*)take((size_t)L_LAYERS * 4 * 2048 * 4);
  unsigned int* wf_c1 = (unsigned int*)take((size_t)L_LAYERS * 4 * 2048 * 4);
  unsigned int* wf_n1 = (unsigned int*)take((size_t)L_LAYERS * 8 * 2048 * 4);
  unsigned int* wf_n2 = (unsigned int*)take((size_t)L_LAYERS * 4 * 2048 * 4);

  // init
  zero_f32<<<(N_NODES * HDIM + 255) / 256, 256, 0, stream>>>(agg_m, (long)N_NODES * HDIM);
  zero_f32<<<(N_NODES * 3 + 255) / 256, 256, 0, stream>>>(agg_x, (long)N_NODES * 3);
  zero_f32<<<(N_NODES + 255) / 256, 256, 0, stream>>>(cnt, (long)N_NODES);
  zero_f32<<<1, 256, 0, stream>>>(hsum, 128);
  copy_f32<<<(N_NODES * 3 + 255) / 256, 256, 0, stream>>>(x_in, x_cur, (long)N_NODES * 3);
  count_kernel<<<(N_EDGES + 255) / 256, 256, 0, stream>>>(edges, cnt);
  embed_kernel<<<(N_NODES * HDIM + 255) / 256, 256, 0, stream>>>(h_in, emb_in_w, emb_in_b, h_bf);

  // weight fragment swizzle (bf16 WMMA B layout)
  swizzle_w<<<(L_LAYERS * 8 * 2048 + 255) / 256, 256, 0, stream>>>(edge_w1, wf_e1, 8, 257 * HDIM);
  swizzle_w<<<(L_LAYERS * 4 * 2048 + 255) / 256, 256, 0, stream>>>(edge_w2, wf_e2, 4, HDIM * HDIM);
  swizzle_w<<<(L_LAYERS * 4 * 2048 + 255) / 256, 256, 0, stream>>>(coord_w1, wf_c1, 4, HDIM * HDIM);
  swizzle_w<<<(L_LAYERS * 8 * 2048 + 255) / 256, 256, 0, stream>>>(node_w1, wf_n1, 8, 2 * HDIM * HDIM);
  swizzle_w<<<(L_LAYERS * 4 * 2048 + 255) / 256, 256, 0, stream>>>(node_w2, wf_n2, 4, HDIM * HDIM);

  // layers
  for (int l = 0; l < L_LAYERS; ++l) {
    edge_kernel<<<N_EDGES / 16, 32, 0, stream>>>(
        edges, h_bf, x_cur, wf_e1, wf_e2, wf_c1, edge_w1, edge_b1, edge_b2,
        att_w, att_b, coord_b1, coord_w2, agg_m, agg_x, l);
    node_kernel<<<N_NODES / 16, 32, 0, stream>>>(
        h_bf, x_cur, agg_m, agg_x, cnt, wf_n1, wf_n2, node_b1, node_b2, l);
  }

  // readout: mean-pool is linear -> sum h, tiny matvec chain
  colsum_kernel<<<250, 128, 0, stream>>>(h_bf, hsum);
  head_kernel<<<1, 128, 0, stream>>>(hsum, emb_out_w, emb_out_b, head_w1, head_b1,
                                     head_w2, head_b2, (float*)d_out);
}
